// AttentionLayer_15994458210508
// MI455X (gfx1250) — matrix operand order repaired and checked by reading, not compiled, not run on hardware
//
#include <hip/hip_runtime.h>
#include <hip/hip_bf16.h>

// DIN attention, fused single-pass kernel for gfx1250 (MI455X).
// B=4096, S=200, H=128. One workgroup per batch row; 8 waves (wave32).
// hist staged to LDS via Tensor Data Mover (TDM) with HW row padding;
// MLP runs as f16 WMMA (f32 accum) with dual accumulators;
// b2 is a constant shift on scores -> cancels in softmax, omitted.

#define BATCH   4096
#define SEQ     200
#define SPAD    208      // 13 tiles of 16 rows
#define HID     128
#define HSTRIDE 132      // LDS row stride in floats = 128 + TDM pad of 4 DWORDs
#define MTILES  13
#define BLOCK   256

typedef __attribute__((ext_vector_type(16))) _Float16     v16h;
typedef __attribute__((ext_vector_type(8)))  float        v8f;
typedef __attribute__((ext_vector_type(4)))  unsigned int v4u;
typedef __attribute__((ext_vector_type(8)))  int          v8i;
typedef __attribute__((ext_vector_type(4)))  int          v4i;

union V16 { v16h v; _Float16 h[16]; };

__global__ __launch_bounds__(BLOCK) void din_attn_kernel(
    const float* __restrict__ hist,   // (B,S,H)
    const float* __restrict__ tgt,    // (B,H)
    const float* __restrict__ mask,   // (B,S)
    const float* __restrict__ W1,     // (3H,H) row-major
    const float* __restrict__ b1,     // (H)
    const float* __restrict__ W2,     // (H,1)
    float* __restrict__ out)          // (B,H)
{
    extern __shared__ float smem[];
    float* sh_hist   = smem;                        // SPAD*HSTRIDE
    float* sh_scores = sh_hist + SPAD * HSTRIDE;    // SPAD
    float* sh_tvec   = sh_scores + SPAD;            // HID
    float* sh_tgt    = sh_tvec + HID;               // HID
    float* sh_red    = sh_tgt + HID;                // 8

    const int tid  = threadIdx.x;
    const int lane = tid & 31;
    const int wave = tid >> 5;
    const int b    = blockIdx.x;

    // ---- TDM: stage hist_b (200x128 f32) into LDS with pad->stride 132 ----
    // Issued once by wave 0 (EXEC is ignored for tensor ops; guard per-wave).
    if (wave == 0) {
        unsigned long long ga =
            (unsigned long long)(uintptr_t)(hist + (size_t)b * SEQ * HID);
        unsigned int ldsa = (unsigned int)(uintptr_t)sh_hist;  // LDS byte offset
        v4u g0;
        g0.x = 1u;                                  // count=1, is_restore=0, no gather
        g0.y = ldsa;                                // lds_addr
        g0.z = (unsigned int)(ga & 0xFFFFFFFFu);    // global_addr[31:0]
        g0.w = (unsigned int)((ga >> 32) & 0x01FFFFFFu) | (2u << 30); // ga[56:32], type=2
        v8i g1;
        g1.s0 = (int)((2u << 16) |                  // data_size = 4B
                      (1u << 20) |                  // pad_enable
                      (6u << 22) |                  // pad_interval: 128 DWORDs
                      (3u << 25));                  // pad_amount: 4 DWORDs
        g1.s1 = (int)(128u << 16);                  // tensor_dim0 = 128 (bits 79:48)
        g1.s2 = (int)(200u << 16);                  // tensor_dim1 = 200 (bits 111:80)
        g1.s3 = (int)(128u << 16);                  // tile_dim0 = 128 (bits 127:112)
        g1.s4 = (int)200u;                          // tile_dim1 = 200
        g1.s5 = (int)128u;                          // tensor_dim0_stride = 128
        g1.s6 = (int)(25600u << 16);                // tensor_dim1_stride[15:0] (unused for 2D)
        g1.s7 = 0;
        v4i gz4 = {0, 0, 0, 0};
        v8i gz8 = {0, 0, 0, 0, 0, 0, 0, 0};
        // clang-23 / therock toolchain: 6-arg form
        __builtin_amdgcn_tensor_load_to_lds(g0, g1, gz4, gz4, gz8, 0);
    }

    if (tid < HID)  sh_tgt[tid] = tgt[(size_t)b * HID + tid];
    if (tid < SPAD) sh_scores[tid] = 0.f;
    __syncthreads();   // sh_tgt / sh_scores ready (TDM still in flight)

    // ---- tvec[n] = b1[n] + sum_k tgt[k] * W1b[k][n]  (overlaps TDM) ----
    if (tid < HID) {
        float acc = b1[tid];
        #pragma unroll 8
        for (int k = 0; k < HID; ++k)
            acc = fmaf(sh_tgt[k], W1[(size_t)(HID + k) * HID + tid], acc);
        sh_tvec[tid] = acc;
    }

    // ---- Per-wave resident B fragments (n-tile = wave), overlaps TDM ----
    // B 32x16 f16 layout per lane: col=lane%16, K = kloA + {0..7} and kloA+16 + {0..7} (same as A)
    const int ncol  = lane & 15;
    const int n     = wave * 16 + ncol;
    const int krowB = (lane < 16) ? 0 : 16;
    // A 16x32 f16 layout per lane: K = kloA + {0..7} and kloA+16 + {0..7}
    const int kloA  = (lane < 16) ? 0 : 8;

    V16 Ba[4], Bc[4], Tg[4];
    #pragma unroll
    for (int kt = 0; kt < 4; ++kt) {
        #pragma unroll
        for (int j = 0; j < 16; ++j) {
            int kb = kt * 32 + kloA + ((j < 8) ? j : (j + 8));
            Ba[kt].h[j] = (_Float16)W1[(size_t)kb * HID + n];             // W1a rows 0..127
            Bc[kt].h[j] = (_Float16)W1[(size_t)(2 * HID + kb) * HID + n]; // W1c rows 256..383
            int ka = kt * 32 + kloA + ((j < 8) ? j : (j + 8));
            Tg[kt].h[j] = (_Float16)sh_tgt[ka];                           // tgt in A-pattern
        }
    }
    const float w2n = W2[n];

    if (wave == 0) __builtin_amdgcn_s_wait_tensorcnt(0);  // hist landed in LDS
    __syncthreads();                                      // visible to all waves
    const float tv = sh_tvec[n];

    // ---- GEMM over 13 M-tiles: 8x v_wmma_f32_16x16x32_f16 per tile (K=256),
    //      dual accumulators to break the WMMA RAW chain ----
    for (int mt = 0; mt < MTILES; ++mt) {
        v8f acc_a = {};
        v8f acc_c = {};
        const float* hrow = &sh_hist[(mt * 16 + ncol) * HSTRIDE];
        #pragma unroll
        for (int kt = 0; kt < 4; ++kt) {
            int kb = kt * 32 + kloA;
            float4 x0 = *(const float4*)&hrow[kb];
            float4 x1 = *(const float4*)&hrow[kb + 4];
            float4 x2 = *(const float4*)&hrow[kb + 16];
            float4 x3 = *(const float4*)&hrow[kb + 20];
            V16 a;
            a.h[0]  = (_Float16)x0.x; a.h[1]  = (_Float16)x0.y;
            a.h[2]  = (_Float16)x0.z; a.h[3]  = (_Float16)x0.w;
            a.h[4]  = (_Float16)x1.x; a.h[5]  = (_Float16)x1.y;
            a.h[6]  = (_Float16)x1.z; a.h[7]  = (_Float16)x1.w;
            a.h[8]  = (_Float16)x2.x; a.h[9]  = (_Float16)x2.y;
            a.h[10] = (_Float16)x2.z; a.h[11] = (_Float16)x2.w;
            a.h[12] = (_Float16)x3.x; a.h[13] = (_Float16)x3.y;
            a.h[14] = (_Float16)x3.z; a.h[15] = (_Float16)x3.w;
            V16 ac; ac.v = a.v * Tg[kt].v;   // hist (.) tgt in f16 (v_pk_mul_f16)
            acc_a = __builtin_amdgcn_wmma_f32_16x16x32_f16(
                false, a.v, false, Ba[kt].v, (short)0, acc_a, false, false);
            acc_c = __builtin_amdgcn_wmma_f32_16x16x32_f16(
                false, ac.v, false, Bc[kt].v, (short)0, acc_c, false, false);
        }
        v8f acc = acc_a + acc_c;
        // relu + dot with W2 segment + reduce over the 16 cols this wave owns
        const int rbase = mt * 16 + ((lane < 16) ? 0 : 8);
        #pragma unroll
        for (int r = 0; r < 8; ++r) {
            float h = acc[r] + tv;
            h = fmaxf(h, 0.f);
            float p = h * w2n;
            p += __shfl_xor(p, 1, 32);
            p += __shfl_xor(p, 2, 32);
            p += __shfl_xor(p, 4, 32);
            p += __shfl_xor(p, 8, 32);
            if (ncol == 0) atomicAdd(&sh_scores[rbase + r], p);  // ds_add_f32
        }
    }
    __syncthreads();

    // ---- mask + pad ----
    if (tid < SPAD) {
        float sc = sh_scores[tid];
        if (tid >= SEQ || mask[(size_t)b * SEQ + tid] < 0.5f) sc = -1.0e9f;
        sh_scores[tid] = sc;
    }
    __syncthreads();

    // ---- softmax over S (wave 0) ----
    if (wave == 0) {
        float m = -3.0e38f;
        for (int s = lane; s < SPAD; s += 32) m = fmaxf(m, sh_scores[s]);
        #pragma unroll
        for (int d = 1; d < 32; d <<= 1) m = fmaxf(m, __shfl_xor(m, d, 32));
        float sum = 0.f;
        for (int s = lane; s < SPAD; s += 32) {
            float e = __expf(sh_scores[s] - m);
            sh_scores[s] = e;
            sum += e;
        }
        #pragma unroll
        for (int d = 1; d < 32; d <<= 1) sum += __shfl_xor(sum, d, 32);
        if (lane == 0) sh_red[0] = 1.0f / sum;
    }
    __syncthreads();

    // ---- out_b[h] = sum_s attn_s * hist[s][h], straight from LDS copy ----
    if (tid < HID) {
        const float inv = sh_red[0];
        float acc = 0.f;
        for (int s = 0; s < SEQ; ++s)
            acc = fmaf(sh_scores[s], sh_hist[s * HSTRIDE + tid], acc);
        out[(size_t)b * HID + tid] = acc * inv;
    }
}

extern "C" void kernel_launch(void* const* d_in, const int* in_sizes, int n_in,
                              void* d_out, int out_size, void* d_ws, size_t ws_size,
                              hipStream_t stream) {
    const float* hist = (const float*)d_in[0];
    const float* tgt  = (const float*)d_in[1];
    const float* mask = (const float*)d_in[2];
    const float* W1   = (const float*)d_in[3];
    const float* b1   = (const float*)d_in[4];
    const float* W2   = (const float*)d_in[5];
    // d_in[6] = b2: constant shift on pre-softmax scores, cancels in softmax.
    float* out = (float*)d_out;

    size_t shbytes = (size_t)(SPAD * HSTRIDE + SPAD + HID + HID + 8) * sizeof(float);
    din_attn_kernel<<<BATCH, BLOCK, shbytes, stream>>>(hist, tgt, mask, W1, b1, W2, out);
}
